// GroupedQueryAttention_26164940767706
// MI455X (gfx1250) — matrix operand order repaired and checked by reading, not compiled, not run on hardware
//
#include <hip/hip_runtime.h>
#include <hip/hip_bf16.h>

// ---------------------------------------------------------------------------
// GQA for MI455X (gfx1250, wave32, WMMA). All matmuls run through one generic
// bf16 WMMA GEMM template (v_wmma_f32_16x16x32_bf16), f32 accumulation.
// Workspace layout (bf16 halfs): Qp[2048x4096] Kp[2048x1024] Vp[2048x1024]
// Xb[2048x4096]  => 40 MiB total.
// d_out layout: output (2048*4096 f32) followed by p_attn (32*2048*2048 f32).
// ---------------------------------------------------------------------------

typedef __attribute__((ext_vector_type(16))) __bf16          v16bf;
typedef __attribute__((ext_vector_type(8)))  float           v8f;
typedef __attribute__((ext_vector_type(16))) unsigned short  u16x16;
typedef __attribute__((ext_vector_type(8)))  unsigned short  u16x8;

__device__ __forceinline__ unsigned short f32_to_bf16(float f) {
    unsigned int u = __builtin_bit_cast(unsigned int, f);
    unsigned int r = u + 0x7FFFu + ((u >> 16) & 1u);   // round-to-nearest-even
    return (unsigned short)(r >> 16);
}

// ---------------------------------------------------------------------------
// Generic tiled WMMA GEMM:  C[z,m,n] = alpha * sum_k A[z,m,k] * B[z,k,n]
//                                       (+ bias[n]) (mask==0 -> -1e4)
// A element (z,m,k): base + z*aHS + m*aRS + k              (k contiguous)
// B element (z,k,n): base + (z/bHDIV)*bHS + k*bSK + n*bSN
//   B_NC=true : bSN==1 (n contiguous)   [weights, V]
//   B_NC=false: bSK==1 (k contiguous)   [K^T for scores]
// Block = 256 threads = 8 waves; block tile 128x128; K step 32.
// Wave (wr,wc) owns a 64x32 sub-tile = 4x2 WMMA 16x16 accumulators.
// HAS_BIAS / HAS_MASK are compile-time so the epilogue is branch-free.
// ---------------------------------------------------------------------------
template <bool A_F32, bool B_F32, bool B_NC, bool C_F32, bool HAS_BIAS, bool HAS_MASK>
__global__ __launch_bounds__(256) void gqa_gemm_wmma(
    const void* __restrict__ aP, long aRS, long aHS,
    const void* __restrict__ bP, long bSK, long bSN, long bHS, int bHDIV,
    void* __restrict__ cP, long cRS, long cHS,
    const float* __restrict__ bias, float alpha,
    const int* __restrict__ mask, long maskRS,
    int M, int N, int K)
{
    constexpr int LDT = 40;                      // padded LDS row (halfs), 16B-aligned
    __shared__ unsigned short As[128 * LDT];     // A tile, row-major (m, k)
    __shared__ unsigned short Bs[128 * LDT];     // B tile, stored transposed (n, k)

    const int tid  = threadIdx.x;
    const int lane = tid & 31;
    const int w    = tid >> 5;
    const int wr   = w >> 2;          // 0..1  (wave row: 64-row half)
    const int wc   = w & 3;           // 0..3  (wave col: 32-col quarter)
    const int bm   = blockIdx.y * 128;
    const int bn   = blockIdx.x * 128;
    const int z    = blockIdx.z;

    const long aOff = (long)z * aHS;
    const long bOff = (long)(z / bHDIV) * bHS;

    v8f acc[4][2];
#pragma unroll
    for (int i = 0; i < 4; ++i)
#pragma unroll
        for (int j = 0; j < 2; ++j)
#pragma unroll
            for (int e = 0; e < 8; ++e) acc[i][j][e] = 0.0f;

    // cooperative-load index maps (16 elements per thread per tile)
    const int aRow = tid >> 1;            // 0..127
    const int aCb  = (tid & 1) * 16;      // 0 | 16
    const int bK   = tid >> 3;            // 0..31   (B_NC path)
    const int bNb  = (tid & 7) * 16;      // 0..112
    const int bN   = tid >> 1;            // 0..127  (!B_NC path)
    const int bKb  = (tid & 1) * 16;      // 0 | 16

    const int r  = lane & 15;             // fragment row/col within 16
    const int kh = lane >> 4;             // lane half selector

    for (int k0 = 0; k0 < K; k0 += 32) {
        // ---- stage A tile (128 x 32) into LDS as bf16 ----
        {
            unsigned short* dst = &As[aRow * LDT + aCb];
            if constexpr (A_F32) {
                const float* src = (const float*)aP + aOff + (long)(bm + aRow) * aRS + (k0 + aCb);
                const float4* s4 = (const float4*)src;
                float4 f0 = s4[0], f1 = s4[1], f2 = s4[2], f3 = s4[3];
                dst[0]  = f32_to_bf16(f0.x); dst[1]  = f32_to_bf16(f0.y);
                dst[2]  = f32_to_bf16(f0.z); dst[3]  = f32_to_bf16(f0.w);
                dst[4]  = f32_to_bf16(f1.x); dst[5]  = f32_to_bf16(f1.y);
                dst[6]  = f32_to_bf16(f1.z); dst[7]  = f32_to_bf16(f1.w);
                dst[8]  = f32_to_bf16(f2.x); dst[9]  = f32_to_bf16(f2.y);
                dst[10] = f32_to_bf16(f2.z); dst[11] = f32_to_bf16(f2.w);
                dst[12] = f32_to_bf16(f3.x); dst[13] = f32_to_bf16(f3.y);
                dst[14] = f32_to_bf16(f3.z); dst[15] = f32_to_bf16(f3.w);
            } else {
                const unsigned short* src =
                    (const unsigned short*)aP + aOff + (long)(bm + aRow) * aRS + (k0 + aCb);
                const uint4* s4 = (const uint4*)src;
                uint4 v0 = s4[0], v1 = s4[1];
                unsigned int* d32 = (unsigned int*)dst;  // 4B aligned (see strides)
                d32[0] = v0.x; d32[1] = v0.y; d32[2] = v0.z; d32[3] = v0.w;
                d32[4] = v1.x; d32[5] = v1.y; d32[6] = v1.z; d32[7] = v1.w;
            }
        }
        // ---- stage B tile (32k x 128n), stored transposed Bs[n][k] ----
        if constexpr (B_NC) {
            if constexpr (B_F32) {
                const float* src = (const float*)bP + bOff + (long)(k0 + bK) * bSK + (bn + bNb);
                const float4* s4 = (const float4*)src;
                float4 f[4] = { s4[0], s4[1], s4[2], s4[3] };
#pragma unroll
                for (int i = 0; i < 4; ++i) {
                    Bs[(bNb + 4 * i + 0) * LDT + bK] = f32_to_bf16(f[i].x);
                    Bs[(bNb + 4 * i + 1) * LDT + bK] = f32_to_bf16(f[i].y);
                    Bs[(bNb + 4 * i + 2) * LDT + bK] = f32_to_bf16(f[i].z);
                    Bs[(bNb + 4 * i + 3) * LDT + bK] = f32_to_bf16(f[i].w);
                }
            } else {
                const unsigned short* src =
                    (const unsigned short*)bP + bOff + (long)(k0 + bK) * bSK + (bn + bNb);
                const uint4* s4 = (const uint4*)src;
                uint4 v0 = s4[0], v1 = s4[1];
                unsigned int u[8] = { v0.x, v0.y, v0.z, v0.w, v1.x, v1.y, v1.z, v1.w };
#pragma unroll
                for (int i = 0; i < 8; ++i) {
                    Bs[(bNb + 2 * i + 0) * LDT + bK] = (unsigned short)(u[i] & 0xFFFFu);
                    Bs[(bNb + 2 * i + 1) * LDT + bK] = (unsigned short)(u[i] >> 16);
                }
            }
        } else {
            if constexpr (B_F32) {
                const float* src = (const float*)bP + bOff + (long)(bn + bN) * bSN + (k0 + bKb);
                const float4* s4 = (const float4*)src;
                float4 f[4] = { s4[0], s4[1], s4[2], s4[3] };
                unsigned short* dstB = &Bs[bN * LDT + bKb];
#pragma unroll
                for (int i = 0; i < 4; ++i) {
                    dstB[4 * i + 0] = f32_to_bf16(f[i].x);
                    dstB[4 * i + 1] = f32_to_bf16(f[i].y);
                    dstB[4 * i + 2] = f32_to_bf16(f[i].z);
                    dstB[4 * i + 3] = f32_to_bf16(f[i].w);
                }
            } else {
                const unsigned short* src =
                    (const unsigned short*)bP + bOff + (long)(bn + bN) * bSN + (k0 + bKb);
                const uint4* s4 = (const uint4*)src;
                uint4 v0 = s4[0], v1 = s4[1];
                unsigned int* d32 = (unsigned int*)&Bs[bN * LDT + bKb];
                d32[0] = v0.x; d32[1] = v0.y; d32[2] = v0.z; d32[3] = v0.w;
                d32[4] = v1.x; d32[5] = v1.y; d32[6] = v1.z; d32[7] = v1.w;
            }
        }

        __syncthreads();

        // ---- fragments (contiguous k per lane half) + 8 WMMAs ----
        v16bf afrag[4], bfrag[2];
#pragma unroll
        for (int mt = 0; mt < 4; ++mt) {
            const int row = wr * 64 + mt * 16 + r;
            // lanes 0-15: K 0-15 ; lanes 16-31: K 16-31
            const u16x8 lo = *(const u16x8*)&As[row * LDT + kh * 16];
            const u16x8 hi = *(const u16x8*)&As[row * LDT + kh * 16 + 8];
            u16x16 t;
#pragma unroll
            for (int i = 0; i < 8; ++i) { t[i] = lo[i]; t[8 + i] = hi[i]; }
            afrag[mt] = __builtin_bit_cast(v16bf, t);
        }
#pragma unroll
        for (int nt = 0; nt < 2; ++nt) {
            const int col = wc * 32 + nt * 16 + r;
            // lane half kh holds K = kh*16 .. kh*16+15 of its column
            const u16x16 t = *(const u16x16*)&Bs[col * LDT + kh * 16];
            bfrag[nt] = __builtin_bit_cast(v16bf, t);
        }
#pragma unroll
        for (int mt = 0; mt < 4; ++mt)
#pragma unroll
            for (int nt = 0; nt < 2; ++nt)
                acc[mt][nt] = __builtin_amdgcn_wmma_f32_16x16x32_bf16(
                    false, afrag[mt], false, bfrag[nt],
                    (short)0, acc[mt][nt], false, false);

        __syncthreads();
    }

    // ---- epilogue: scale, bias, mask, store (f32 or bf16), branch-free ----
#pragma unroll
    for (int nt = 0; nt < 2; ++nt) {
        const int n = bn + wc * 32 + nt * 16 + r;       // fixed per lane per nt
        float bn_val = 0.0f;
        if constexpr (HAS_BIAS) bn_val = bias[n];       // one load per (nt)
#pragma unroll
        for (int mt = 0; mt < 4; ++mt) {
            const int  mBase = bm + wr * 64 + mt * 16 + kh * 8;   // D-matrix layout
            const long cBase = (long)z * cHS + (long)mBase * cRS + n;
#pragma unroll
            for (int v = 0; v < 8; ++v) {
                float val = acc[mt][nt][v] * alpha;
                if constexpr (HAS_BIAS) val += bn_val;
                if constexpr (HAS_MASK) {
                    if (mask[(long)(mBase + v) * maskRS + n] == 0) val = -10000.0f;
                }
                const long cIdx = cBase + (long)v * cRS;
                if constexpr (C_F32) ((float*)cP)[cIdx] = val;
                else ((unsigned short*)cP)[cIdx] = f32_to_bf16(val);
            }
        }
    }
}

// ---------------------------------------------------------------------------
// Single-pass row softmax: one block per (h,q) row; 8 elements per thread
// held in registers (one HBM read + one write of the 536 MB p_attn).
// ---------------------------------------------------------------------------
__global__ __launch_bounds__(256) void gqa_softmax_rows(float* __restrict__ p, int S)
{
    const long row = blockIdx.x;
    float* x = p + row * (long)S;
    const int tid = threadIdx.x;
    __shared__ float red[256];

    float vals[8];
    float mx = -3.0e38f;
#pragma unroll
    for (int i = 0; i < 8; ++i) {
        vals[i] = x[tid + i * 256];
        mx = fmaxf(mx, vals[i]);
    }
    red[tid] = mx; __syncthreads();
    for (int s = 128; s > 0; s >>= 1) {
        if (tid < s) red[tid] = fmaxf(red[tid], red[tid + s]);
        __syncthreads();
    }
    mx = red[0]; __syncthreads();

    float sum = 0.0f;
#pragma unroll
    for (int i = 0; i < 8; ++i) {
        vals[i] = __expf(vals[i] - mx);
        sum += vals[i];
    }
    red[tid] = sum; __syncthreads();
    for (int s = 128; s > 0; s >>= 1) {
        if (tid < s) red[tid] += red[tid + s];
        __syncthreads();
    }
    const float inv = 1.0f / red[0];
#pragma unroll
    for (int i = 0; i < 8; ++i) x[tid + i * 256] = vals[i] * inv;
}

// ---------------------------------------------------------------------------
extern "C" void kernel_launch(void* const* d_in, const int* in_sizes, int n_in,
                              void* d_out, int out_size, void* d_ws, size_t ws_size,
                              hipStream_t stream)
{
    (void)in_sizes; (void)n_in; (void)out_size; (void)ws_size;

    constexpr int  S  = 2048, D = 4096, H = 32, KV = 8, DK = 128, G = 4;
    constexpr int  NKV = KV * DK;                 // 1024
    constexpr long SS  = (long)S * S;

    const float* query = (const float*)d_in[0];
    const float* key   = (const float*)d_in[1];
    const float* value = (const float*)d_in[2];
    const int*   mask  = (const int*)  d_in[3];
    const float* Wq    = (const float*)d_in[4];
    const float* bq    = (const float*)d_in[5];
    const float* Wk    = (const float*)d_in[6];
    const float* bk    = (const float*)d_in[7];
    const float* Wv    = (const float*)d_in[8];
    const float* bv    = (const float*)d_in[9];
    const float* Wo    = (const float*)d_in[10];
    const float* bo    = (const float*)d_in[11];

    // workspace (bf16 halfs): Qp | Kp | Vp | Xb  = 40 MiB
    unsigned short* Qp = (unsigned short*)d_ws;
    unsigned short* Kp = Qp + (long)S * D;
    unsigned short* Vp = Kp + (long)S * NKV;
    unsigned short* Xb = Vp + (long)S * NKV;

    float* out    = (float*)d_out;                // (S, D)
    float* p_attn = out + (long)S * D;            // (H, S, S)

    const float alpha_s = 0.08838834764831843f;   // 1/sqrt(128)

    // 1) Qp = bf16(query @ Wq + bq)          M=2048 N=4096 K=4096
    gqa_gemm_wmma<true, true, true, false, true, false>
        <<<dim3(D / 128, S / 128, 1), 256, 0, stream>>>(
        query, D, 0,  Wq, D, 1, 0, 1,  Qp, D, 0,  bq, 1.0f, nullptr, 0, S, D, D);

    // 2) Kp = bf16(key @ Wk + bk)            M=2048 N=1024 K=4096
    gqa_gemm_wmma<true, true, true, false, true, false>
        <<<dim3(NKV / 128, S / 128, 1), 256, 0, stream>>>(
        key, D, 0,  Wk, NKV, 1, 0, 1,  Kp, NKV, 0,  bk, 1.0f, nullptr, 0, S, NKV, D);

    // 3) Vp = bf16(value @ Wv + bv)
    gqa_gemm_wmma<true, true, true, false, true, false>
        <<<dim3(NKV / 128, S / 128, 1), 256, 0, stream>>>(
        value, D, 0,  Wv, NKV, 1, 0, 1,  Vp, NKV, 0,  bv, 1.0f, nullptr, 0, S, NKV, D);

    // 4) scores[h] = mask(alpha * Qp[:,h] @ Kp[:,h/G]^T)  -> p_attn (f32)
    //    per head: M=2048 N=2048 K=128 ; B(k,n)=Kp[n, (h/G)*DK + k]  (k contiguous)
    gqa_gemm_wmma<false, false, false, true, false, true>
        <<<dim3(S / 128, S / 128, H), 256, 0, stream>>>(
        Qp, D, DK,  Kp, 1, NKV, DK, G,  p_attn, S, SS,
        nullptr, alpha_s, mask, S, S, S, DK);

    // 5) row softmax in place on p_attn (H*S rows of length S)
    gqa_softmax_rows<<<H * S, 256, 0, stream>>>(p_attn, S);

    // 6) Xb[:, h*DK: ] = bf16(p_attn[h] @ Vp[:, (h/G)*DK: ])   M=2048 N=128 K=2048
    gqa_gemm_wmma<true, false, true, false, false, false>
        <<<dim3(DK / 128, S / 128, H), 256, 0, stream>>>(
        p_attn, S, SS,  Vp, NKV, 1, DK, G,  Xb, D, DK,
        nullptr, 1.0f, nullptr, 0, S, DK, S);

    // 7) out = Xb @ Wo + bo                   M=2048 N=4096 K=4096
    gqa_gemm_wmma<false, true, true, true, true, false>
        <<<dim3(D / 128, S / 128, 1), 256, 0, stream>>>(
        Xb, D, 0,  Wo, D, 1, 0, 1,  out, D, 0,  bo, 1.0f, nullptr, 0, S, D, D);
}
